// MultiTaskGNNModel_40733469835807
// MI455X (gfx1250) — matrix operand order repaired and checked by reading, not compiled, not run on hardware
//
#include <hip/hip_runtime.h>

// ============================================================================
// MultiTaskGNNModel on MI455X (gfx1250, wave32).
//
// Live computation (relation segment-sums are dead code): per-row MLP
// 32->64->64->64 (+ReLU) + three softmax heads (21/2/5) -> [N,28].
// Memory floor ~10us (240 MB @ 23.3 TB/s); the kernel is instruction-bound,
// so v4 keeps shaving the per-row stream:
//  - transposed chain H^T = W^T @ H^T (ind in lanes for both B and D):
//    B = contiguous v16h LDS load, D = packed v8h store, pk_max f16 ReLU
//  - ALL weights (incl. fused heads) pre-swizzled into A-fragment layout in
//    LDS by the block; waves fetch fragments as one 32B LDS read
//  - head logits overlay the dead H tile (stride-28 rows): no Ss buffer,
//    division-free coalesced out-store
//  - dual-tile softmax: lanes 0-15 do tile t, lanes 16-31 tile t+1 (no
//    divergence, 2x softmax throughput)
// ============================================================================

typedef __attribute__((ext_vector_type(16))) _Float16 v16h;
typedef __attribute__((ext_vector_type(8)))  _Float16 v8h;
typedef __attribute__((ext_vector_type(2)))  _Float16 v2h;
typedef __attribute__((ext_vector_type(8)))  float    v8f;
typedef __attribute__((ext_vector_type(4)))  float    v4f;

#define WAVES          4
#define TILES          4
#define ROWS_PER_WAVE  (TILES * 16)
#define ROWS_PER_BLOCK (WAVES * ROWS_PER_WAVE) // 256
#define HID            64
#define NHEAD          28
#define NFRAG          24                      // L1:4  L2:8  L3:8  HEAD:4
#define FB_L1          0
#define FB_L2          4
#define FB_L3          12
#define FB_HD          20

__device__ __forceinline__ void wait_lds() {
  asm volatile("s_wait_dscnt 0" ::: "memory");   // gfx1250 split counter
}

__device__ __forceinline__ v8f wmma_f16(v16h a, v16h b, v8f c) {
  return __builtin_amdgcn_wmma_f32_16x16x32_f16(
      false, a, false, b, (short)0, c, false, false);
}

// --- Fused head lookup: class cols [0,21)=age, [21,23)=sex, [23,28)=eth.
__device__ __forceinline__ float head_w(const float* Wa, const float* Ws,
                                        const float* We, int k, int col) {
  if (col < 21) return Wa[k * 21 + col];
  if (col < 23) return Ws[k * 2 + (col - 21)];
  if (col < 28) return We[k * 5 + (col - 23)];
  return 0.0f;
}
__device__ __forceinline__ float head_b(const float* ba, const float* bs,
                                        const float* be, int col) {
  if (col < 21) return ba[col];
  if (col < 23) return bs[col - 21];
  if (col < 28) return be[col - 23];
  return 0.0f;
}

// ---------------------------------------------------------------------------
// Block-cooperative weight staging in A-fragment layout Wf[frag][lane][16].
// A layout: lane M = lane&15, element i -> K = kc*32 + laneHalf*16 + i.
// Each thread owns two fixed (lane',even-i) slots;
// adjacent i = adjacent K -> 2 loads + cvt_pk + ds_store_b32 per fragment.
// ---------------------------------------------------------------------------
__device__ __forceinline__ void fill_wfrags(const float* __restrict__ W1,
                                            const float* __restrict__ W2,
                                            const float* __restrict__ W3,
                                            const float* __restrict__ Wha,
                                            const float* __restrict__ Whs,
                                            const float* __restrict__ Whe,
                                            _Float16* Wf, int tid) {
#pragma unroll
  for (int s = 0; s < 2; ++s) {
    const int p     = tid + s * (WAVES * 32);   // 0..255
    const int lane_ = p >> 3;                   // 0..31
    const int i0    = (p & 7) << 1;             // even element index 0..14
    const int lh    = lane_ >> 4;
    const int kloc  = (lh << 4) + i0;
    const int src0  = kloc * HID + (lane_ & 15);   // (K,m) offset within W
    const int dst0  = lane_ * 16 + i0;             // halves, within fragment

    int f = 0;
    // layer 1: W1[32][64], kc = 0
#pragma unroll
    for (int mt = 0; mt < 4; ++mt, ++f) {
      const float w0 = W1[src0 + mt * 16];
      const float w1 = W1[src0 + mt * 16 + HID];   // K+1
      v2h h; h[0] = (_Float16)w0; h[1] = (_Float16)w1;
      *(v2h*)(Wf + f * 512 + dst0) = h;
    }
    // layers 2,3: W[64][64], mt x kc
    const float* Ws2[2] = {W2, W3};
#pragma unroll
    for (int l = 0; l < 2; ++l)
#pragma unroll
      for (int mt = 0; mt < 4; ++mt)
#pragma unroll
        for (int kc = 0; kc < 2; ++kc, ++f) {
          const int so = src0 + kc * 32 * HID + mt * 16;
          const float w0 = Ws2[l][so];
          const float w1 = Ws2[l][so + HID];
          v2h h; h[0] = (_Float16)w0; h[1] = (_Float16)w1;
          *(v2h*)(Wf + f * 512 + dst0) = h;
        }
    // fused head [64][28->32]: mt x kc (branch on col is fixed per slot)
#pragma unroll
    for (int mt = 0; mt < 2; ++mt)
#pragma unroll
      for (int kc = 0; kc < 2; ++kc, ++f) {
        const int col = mt * 16 + (lane_ & 15);
        const int K0  = kc * 32 + kloc;
        v2h h;
        h[0] = (_Float16)head_w(Wha, Whs, Whe, K0,     col);
        h[1] = (_Float16)head_w(Wha, Whs, Whe, K0 + 1, col);
        *(v2h*)(Wf + f * 512 + dst0) = h;
      }
  }
}

__device__ __forceinline__ v16h get_wfrag(const _Float16* Wf, int f, int lane) {
  return *(const v16h*)(Wf + f * 512 + lane * 16);   // 2x ds_load_b128
}

// --- B fragment = 32x16 slab of H^T from LDS row L[ind][hid]: contiguous.
__device__ __forceinline__ v16h load_bfrag_lds(const _Float16* Lt, int kc,
                                               int lane) {
  const _Float16* p = Lt + (lane & 15) * HID + kc * 32 + ((lane >> 4) << 4);
  return *(const v16h*)p;
}

// --- Layer-1 B fragment straight from global x[N][32].
__device__ __forceinline__ v16h load_bfrag_x(const float* __restrict__ x,
                                             int ind, int lane) {
  const float* p = x + ind * 32 + ((lane >> 4) << 4);
  const v8f lo = *(const v8f*)(p + 0);
  const v8f hi = *(const v8f*)(p + 8);
  v16h b;
#pragma unroll
  for (int i = 0; i < 8; ++i) b[i] = (_Float16)lo[i];
#pragma unroll
  for (int i = 0; i < 8; ++i) b[8 + i] = (_Float16)hi[i];
  return b;
}

// --- D tile store: cvt_pk to f16, ReLU on packed halves, 1x ds_store_b128.
__device__ __forceinline__ void store_dt(_Float16* Lt, int mt, v8f d, int lane) {
  _Float16* p = Lt + (lane & 15) * HID + mt * 16 + ((lane >> 4) << 3);
  v8h h;
#pragma unroll
  for (int r = 0; r < 8; ++r) h[r] = (_Float16)d[r];
#if __has_builtin(__builtin_elementwise_max)
  v8h z = {};
  h = __builtin_elementwise_max(h, z);
#else
#pragma unroll
  for (int r = 0; r < 8; ++r) h[r] = h[r] > (_Float16)0.0f ? h[r] : (_Float16)0.0f;
#endif
  *(v8h*)p = h;
}

// --- 64->64 dense layer (+bias, ReLU), transposed, in-place in LDS.
__device__ __forceinline__ void dense64T(const _Float16* Wf, int fbase,
                                         const float* __restrict__ bias,
                                         _Float16* Lw, int lane) {
  v16h a[8];
#pragma unroll
  for (int q = 0; q < 8; ++q) a[q] = get_wfrag(Wf, fbase + q, lane);

  v8f cinit[4];
#pragma unroll
  for (int mt = 0; mt < 4; ++mt)
    cinit[mt] = *(const v8f*)(bias + mt * 16 + ((lane >> 4) << 3));

#pragma unroll
  for (int t = 0; t < TILES; ++t) {
    _Float16* Lt = Lw + t * 16 * HID;
    const v16h b0 = load_bfrag_lds(Lt, 0, lane);  // DS in-order: reads precede
    const v16h b1 = load_bfrag_lds(Lt, 1, lane);  // the overwriting stores
#pragma unroll
    for (int mt = 0; mt < 4; ++mt) {
      v8f d = wmma_f16(a[mt * 2 + 0], b0, cinit[mt]);
      d     = wmma_f16(a[mt * 2 + 1], b1, d);
      store_dt(Lt, mt, d, lane);
    }
  }
}

__device__ __forceinline__ void softmax_seg(float* row, int s, int e) {
  float m = -3.402823466e38f;
  for (int j = s; j < e; ++j) m = fmaxf(m, row[j]);
  float sum = 0.0f;
  for (int j = s; j < e; ++j) { float v = __expf(row[j] - m); row[j] = v; sum += v; }
  const float inv = 1.0f / sum;
  for (int j = s; j < e; ++j) row[j] *= inv;
}

__global__ __launch_bounds__(WAVES * 32)
void gnn_mlp_softmax_kernel(const float* __restrict__ x,
                            const float* __restrict__ W1, const float* __restrict__ b1,
                            const float* __restrict__ W2, const float* __restrict__ b2,
                            const float* __restrict__ W3, const float* __restrict__ b3,
                            const float* __restrict__ Wha, const float* __restrict__ bha,
                            const float* __restrict__ Whs, const float* __restrict__ bhs,
                            const float* __restrict__ Whe, const float* __restrict__ bhe,
                            float* __restrict__ out, int Ntot) {
  __shared__ __align__(32) _Float16 Wf[NFRAG * 32 * 16];       // 24 KB weights
  __shared__ __align__(32) _Float16 Hs[WAVES][TILES][16][HID]; // 32 KB H^T
  // head logits overlay dead H tiles; no separate buffer (56 KB total LDS)

  const int tid  = threadIdx.x;
  const int lane = tid & 31;
  const int wv   = tid >> 5;
  const int wave_row0 = blockIdx.x * ROWS_PER_BLOCK + wv * ROWS_PER_WAVE;

  fill_wfrags(W1, W2, W3, Wha, Whs, Whe, Wf, tid);
  __syncthreads();

  _Float16* Lw = &Hs[wv][0][0][0];

  // ---------- Layer 1: W1^T @ x^T (+b1, ReLU) ----------
  {
    v16h a[4];
#pragma unroll
    for (int mt = 0; mt < 4; ++mt) a[mt] = get_wfrag(Wf, FB_L1 + mt, lane);
    v8f cinit[4];
#pragma unroll
    for (int mt = 0; mt < 4; ++mt)
      cinit[mt] = *(const v8f*)(b1 + mt * 16 + ((lane >> 4) << 3));

#pragma unroll
    for (int t = 0; t < TILES; ++t) {
      int ind = wave_row0 + t * 16 + (lane & 15);
      ind = ind < Ntot ? ind : Ntot - 1;          // clamp; out-stores guarded
      const v16h b = load_bfrag_x(x, ind, lane);
      _Float16* Lt = Lw + t * 16 * HID;
#pragma unroll
      for (int mt = 0; mt < 4; ++mt) {
        v8f d = wmma_f16(a[mt], b, cinit[mt]);
        store_dt(Lt, mt, d, lane);
      }
    }
  }
  wait_lds();

  // ---------- Layers 2 & 3 ----------
  dense64T(Wf, FB_L2, b2, Lw, lane);
  wait_lds();
  dense64T(Wf, FB_L3, b3, Lw, lane);
  wait_lds();

  // ---------- Heads + dual-tile softmax ----------
  {
    v16h a[4];
    v8f  cinit[2];
    const int kh = (lane >> 4) << 3;
#pragma unroll
    for (int q = 0; q < 4; ++q) a[q] = get_wfrag(Wf, FB_HD + q, lane);
#pragma unroll
    for (int mt = 0; mt < 2; ++mt)
#pragma unroll
      for (int r = 0; r < 8; ++r)
        cinit[mt][r] = head_b(bha, bhs, bhe, mt * 16 + kh + r);

#pragma unroll
    for (int tp = 0; tp < TILES; tp += 2) {
      // --- head matmuls for tiles tp, tp+1; logits overlay each H tile as
      // float rows of stride NHEAD.  mt=1 stored FIRST: its tail (classes
      // 28-31) spills into the next row's cols 0-3 and is repaired by mt=0.
#pragma unroll
      for (int u = 0; u < 2; ++u) {
        _Float16* Lt = Lw + (tp + u) * 16 * HID;
        const v16h b0 = load_bfrag_lds(Lt, 0, lane);
        const v16h b1 = load_bfrag_lds(Lt, 1, lane);
        float* St = (float*)Lt;
#pragma unroll
        for (int mt = 1; mt >= 0; --mt) {
          v8f d = wmma_f16(a[mt * 2 + 0], b0, cinit[mt]);
          d     = wmma_f16(a[mt * 2 + 1], b1, d);
          float* q = St + (lane & 15) * NHEAD + mt * 16 + kh;
          v4f lo; v4f hi;
#pragma unroll
          for (int r = 0; r < 4; ++r) { lo[r] = d[r]; hi[r] = d[4 + r]; }
          *(v4f*)(q + 0) = lo;                 // 16B-aligned ds_store_b128
          *(v4f*)(q + 4) = hi;
        }
      }
      wait_lds();

      // --- dual-tile softmax: lanes 0-15 tile tp, lanes 16-31 tile tp+1.
      {
        float* row = (float*)(Lw + (tp + (lane >> 4)) * 16 * HID)
                     + (lane & 15) * NHEAD;
        softmax_seg(row,  0, 21);
        softmax_seg(row, 21, 23);
        softmax_seg(row, 23, 28);
      }
      wait_lds();

      // --- coalesced, division-free store of 896 contiguous floats.
      const int row0 = wave_row0 + tp * 16;
      const float* S0 = (const float*)(Lw + tp * 16 * HID);
      const float* S1 = (const float*)(Lw + (tp + 1) * 16 * HID);
      if (row0 + 32 <= Ntot) {
        float* o = out + row0 * NHEAD + lane;
#pragma unroll
        for (int j = 0; j < 14; ++j) o[j * 32]       = S0[lane + j * 32];
#pragma unroll
        for (int j = 0; j < 14; ++j) o[448 + j * 32] = S1[lane + j * 32];
      } else {
        for (int idx = lane; idx < 448; idx += 32) {
          if (row0 + idx / NHEAD < Ntot)
            out[row0 * NHEAD + idx] = S0[idx];
        }
        for (int idx = lane; idx < 448; idx += 32) {
          if (row0 + 16 + idx / NHEAD < Ntot)
            out[(row0 + 16) * NHEAD + idx] = S1[idx];
        }
      }
    }
  }
}

// ============================================================================
// Host launcher.  Input flattening (setup_inputs dict order, leaves in order):
//  0:age_dst 1:sex_dst 2:eth_dst  3..6: emb(individual,age,sex,eth)
//  p1: 7..14 (self_loop W=13,b=14)  p2: 15..22 (W=21,b=22)
//  p3: 23..30 (W=29,b=30)  heads: 31/32 age  33/34 sex  35/36 eth
// ============================================================================
extern "C" void kernel_launch(void* const* d_in, const int* in_sizes, int n_in,
                              void* d_out, int out_size, void* d_ws, size_t ws_size,
                              hipStream_t stream) {
  (void)n_in; (void)out_size; (void)d_ws; (void)ws_size;

  const float* x   = (const float*)d_in[3];
  const float* W1  = (const float*)d_in[13];
  const float* b1  = (const float*)d_in[14];
  const float* W2  = (const float*)d_in[21];
  const float* b2  = (const float*)d_in[22];
  const float* W3  = (const float*)d_in[29];
  const float* b3  = (const float*)d_in[30];
  const float* Wha = (const float*)d_in[31];
  const float* bha = (const float*)d_in[32];
  const float* Whs = (const float*)d_in[33];
  const float* bhs = (const float*)d_in[34];
  const float* Whe = (const float*)d_in[35];
  const float* bhe = (const float*)d_in[36];
  float* out = (float*)d_out;

  const int N      = in_sizes[0];
  const int blocks = (N + ROWS_PER_BLOCK - 1) / ROWS_PER_BLOCK;

  gnn_mlp_softmax_kernel<<<dim3(blocks), dim3(WAVES * 32), 0, stream>>>(
      x, W1, b1, W2, b2, W3, b3, Wha, bha, Whs, bhs, Whe, bhe, out, N);
}
